// NeSVoR_17824114278641
// MI455X (gfx1250) — matrix operand order repaired and checked by reading, not compiled, not run on hardware
//
#include <hip/hip_runtime.h>
#include <hip/hip_bf16.h>
#include <math.h>

// ---------------------------------------------------------------------------
// NeSVoR deformation field: hash-grid encoding + tiny MLP (32->64->64->3)
// MI455X / gfx1250, wave32, WMMA f16 (f32 accum) for the MLP.
// tanh via gfx1250 hardware v_tanh_f32 (branch-free, TRANS pipe).
// Loop order nt{mt}: each B fragment loaded once; back-to-back independent
// WMMAs hide the WMMA->VALU hazard NOPs.
// ---------------------------------------------------------------------------

typedef __attribute__((ext_vector_type(16))) _Float16 v16h;
typedef __attribute__((ext_vector_type(8)))  _Float16 v8h;
typedef __attribute__((ext_vector_type(8)))  float    v8f;

#define LVL   8
#define LOG2T 19
#define TSZ   (1u << LOG2T)
#define DIN   32      // L*F (16) + E_DIM (16)
#define HID   64
#define NWAVE 8       // 256 threads / 32

// Branch-free tanh. gfx1250 has a hardware transcendental v_tanh_f32.
__device__ __forceinline__ float fast_tanh(float v) {
#if __has_builtin(__builtin_amdgcn_tanhf)
  return __builtin_amdgcn_tanhf(v);
#else
  const float xc = fminf(fmaxf(v, -10.0f), 10.0f);
  const float t  = __builtin_amdgcn_exp2f(xc * 2.885390082f);  // exp(2x)
  return (t - 1.0f) * __builtin_amdgcn_rcpf(t + 1.0f);
#endif
}

__device__ __forceinline__ v16h make_frag(v8h lo, v8h hi) {
  v16h r;
#pragma unroll
  for (int i = 0; i < 8; ++i) { r[i] = lo[i]; r[8 + i] = hi[i]; }
  return r;
}

// A-fragment (16x32 f16) from a row-major [rows][C] LDS buffer.
// ISA layout: lanes 0-15 hold K 0..7 then 16..23; lanes 16-31 hold K 8..15 then 24..31.
__device__ __forceinline__ v16h load_a(const _Float16* buf, int C, int row,
                                       int kbase, int lane) {
  const int kh = (lane >> 4) * 8;
  const v8h* p0 = reinterpret_cast<const v8h*>(buf + row * C + kbase + kh);
  const v8h* p1 = reinterpret_cast<const v8h*>(buf + row * C + kbase + 16 + kh);
  return make_frag(*p0, *p1);
}

// B-fragment (32x16 f16) from weights stored [n][K] (row stride KS halfs).
// ISA layout: lane = column (N = lane&15); lanes 0-15 hold K 0..7 then 16..23, lanes 16-31 K 8..15 then 24..31.
__device__ __forceinline__ v16h load_b(const _Float16* w, int KS, int ncol,
                                       int kbase, int lane) {
  const int n  = ncol + (lane & 15);
  const int kb = kbase + (lane >> 4) * 8;
  const v8h* p0 = reinterpret_cast<const v8h*>(w + n * KS + kb);
  const v8h* p1 = reinterpret_cast<const v8h*>(w + n * KS + kb + 16);
  return make_frag(*p0, *p1);
}

// tanh the 8 accumulator elements first (independent TRANS ops pipeline
// without hazard NOPs), then convert+store to the [32][HID] LDS buffer.
__device__ __forceinline__ void tanh_store_tile(_Float16* Hbuf, const v8f acc,
                                                int mt, int nt, int mhi, int nlo) {
  float tv[8];
#pragma unroll
  for (int rr = 0; rr < 8; ++rr) tv[rr] = fast_tanh(acc[rr]);
#pragma unroll
  for (int rr = 0; rr < 8; ++rr) {
    const int row = mt * 16 + mhi + rr;   // C layout: M = vgpr + 8*(lane>=16)
    Hbuf[row * HID + nt * 16 + nlo] = (_Float16)tv[rr];
  }
}

__device__ __forceinline__ float smoothstep_w(float u) {
  return u * u * (3.0f - 2.0f * u);
}

__global__ __launch_bounds__(256) void nesvor_deform(
    const float* __restrict__ x, const float* __restrict__ e,
    const float* __restrict__ tables,
    const float* __restrict__ W1, const float* __restrict__ b1,
    const float* __restrict__ W2, const float* __restrict__ b2,
    const float* __restrict__ W3, const float* __restrict__ b3,
    float* __restrict__ out) {
  // ---- LDS ----------------------------------------------------------------
  __shared__ _Float16 sW1[HID][DIN];    // W1^T  [n][k]  (f16)
  __shared__ _Float16 sW2[HID][HID];    // W2^T  [n][k]  (f16)
  __shared__ float    sW3[HID * 3];     // W3    [k][j]  (f32)
  __shared__ float    sB1[HID], sB2[HID], sB3[4];
  __shared__ _Float16 sA[NWAVE][32][DIN];   // per-wave input rows
  __shared__ _Float16 sH[NWAVE][32][HID];   // per-wave hidden activations

  const int tid  = threadIdx.x;
  const int lane = tid & 31;
  const int wv   = tid >> 5;

  // ---- stage weights once per block (f32 -> f16, transpose to [n][k]) -----
  for (int i = tid; i < DIN * HID; i += 256)   // W1 is [k=32][n=64] row-major
    sW1[i & 63][i >> 6] = (_Float16)W1[i];
  for (int i = tid; i < HID * HID; i += 256)   // W2 is [k=64][n=64] row-major
    sW2[i & 63][i >> 6] = (_Float16)W2[i];
  for (int i = tid; i < HID * 3; i += 256) sW3[i] = W3[i];
  if (tid < HID) { sB1[tid] = b1[tid]; sB2[tid] = b2[tid]; }
  if (tid < 3) sB3[tid] = b3[tid];
  __syncthreads();

  // one point per thread; N is a multiple of 256 so EXEC stays all-ones (WMMA req.)
  const int p = blockIdx.x * 256 + tid;
  const float x0 = x[p * 3 + 0], x1 = x[p * 3 + 1], x2 = x[p * 3 + 2];

  // ---- hash-grid encoding (per lane; 64 random b64 gathers, L2-resident) --
  _Float16 encv[16];
  float rf = 16.0f;  // BASE * SCALE^lev, folded by full unroll
#pragma unroll
  for (int lev = 0; lev < LVL; ++lev) {
    const float r = floorf(rf);
    rf *= 1.5f;
    const float px = x0 * r, py = x1 * r, pz = x2 * r;
    const float fx = floorf(px), fy = floorf(py), fz = floorf(pz);
    const unsigned ix = (unsigned)fx, iy = (unsigned)fy, iz = (unsigned)fz;
    const float tx = smoothstep_w(px - fx);
    const float ty = smoothstep_w(py - fy);
    const float tz = smoothstep_w(pz - fz);
    const float2* tab = reinterpret_cast<const float2*>(tables) + (size_t)lev * TSZ;
    float f0 = 0.0f, f1 = 0.0f;
#pragma unroll
    for (int c = 0; c < 8; ++c) {
      const unsigned cx = ix + ((c >> 2) & 1);
      const unsigned cy = iy + ((c >> 1) & 1);
      const unsigned cz = iz + (c & 1);
      unsigned h = cx ^ (cy * 2654435761u) ^ (cz * 805459861u);
      h &= (TSZ - 1u);
      const float2 f = tab[h];                         // global_load_b64
      const float w = ((c & 4) ? tx : 1.0f - tx) *
                      ((c & 2) ? ty : 1.0f - ty) *
                      ((c & 1) ? tz : 1.0f - tz);
      f0 += w * f.x;
      f1 += w * f.y;
    }
    encv[2 * lev + 0] = (_Float16)f0;
    encv[2 * lev + 1] = (_Float16)f1;
  }

  // ---- stage input row [enc | e] (32 f16) into per-wave A buffer ----------
  {
    v8h c0, c1, c2, c3;
#pragma unroll
    for (int i = 0; i < 8; ++i) { c0[i] = encv[i]; c1[i] = encv[8 + i]; }
    const float4* ep = reinterpret_cast<const float4*>(e + (size_t)p * 16);
    const float4 e0 = ep[0], e1 = ep[1], e2 = ep[2], e3 = ep[3];
    c2[0] = (_Float16)e0.x; c2[1] = (_Float16)e0.y; c2[2] = (_Float16)e0.z; c2[3] = (_Float16)e0.w;
    c2[4] = (_Float16)e1.x; c2[5] = (_Float16)e1.y; c2[6] = (_Float16)e1.z; c2[7] = (_Float16)e1.w;
    c3[0] = (_Float16)e2.x; c3[1] = (_Float16)e2.y; c3[2] = (_Float16)e2.z; c3[3] = (_Float16)e2.w;
    c3[4] = (_Float16)e3.x; c3[5] = (_Float16)e3.y; c3[6] = (_Float16)e3.z; c3[7] = (_Float16)e3.w;
    v8h* dst = reinterpret_cast<v8h*>(&sA[wv][lane][0]);
    dst[0] = c0; dst[1] = c1; dst[2] = c2; dst[3] = c3;
  }
  // same-wave LDS ops are in-order: no barrier needed before reading sA/sH back.

  const _Float16* Abuf = &sA[wv][0][0];
  _Float16*       Hbuf = &sH[wv][0][0];
  const int nlo = lane & 15;
  const int mhi = (lane >> 4) * 8;

  // ---- layer 1: [32x32] @ [32x64] + b1, tanh  (8 WMMAs / wave) ------------
  {
    v16h aF[2];
#pragma unroll
    for (int mt = 0; mt < 2; ++mt)
      aF[mt] = load_a(Abuf, DIN, mt * 16 + nlo, 0, lane);
#pragma unroll
    for (int nt = 0; nt < 4; ++nt) {
      const v16h bf = load_b(&sW1[0][0], DIN, nt * 16, 0, lane);
      const float bv = sB1[nt * 16 + nlo];
#pragma unroll
      for (int mt = 0; mt < 2; ++mt) {    // two independent WMMAs on same B
        v8f acc = {bv, bv, bv, bv, bv, bv, bv, bv};
        acc = __builtin_amdgcn_wmma_f32_16x16x32_f16(
            false, aF[mt], false, bf, (short)0, acc, false, false);
        tanh_store_tile(Hbuf, acc, mt, nt, mhi, nlo);
      }
    }
  }

  // ---- layer 2: [32x64] @ [64x64] + b2, tanh  (16 WMMAs / wave) -----------
  {
    v16h aK0[2], aK1[2];
#pragma unroll
    for (int mt = 0; mt < 2; ++mt) {      // all A reads precede all stores
      aK0[mt] = load_a(Hbuf, HID, mt * 16 + nlo, 0, lane);
      aK1[mt] = load_a(Hbuf, HID, mt * 16 + nlo, 32, lane);
    }
#pragma unroll
    for (int nt = 0; nt < 4; ++nt) {
      const v16h bk0 = load_b(&sW2[0][0], HID, nt * 16, 0, lane);
      const v16h bk1 = load_b(&sW2[0][0], HID, nt * 16, 32, lane);
      const float bv = sB2[nt * 16 + nlo];
#pragma unroll
      for (int mt = 0; mt < 2; ++mt) {
        v8f acc = {bv, bv, bv, bv, bv, bv, bv, bv};
        acc = __builtin_amdgcn_wmma_f32_16x16x32_f16(
            false, aK0[mt], false, bk0, (short)0, acc, false, false);
        acc = __builtin_amdgcn_wmma_f32_16x16x32_f16(
            false, aK1[mt], false, bk1, (short)0, acc, false, false);
        tanh_store_tile(Hbuf, acc, mt, nt, mhi, nlo);
      }
    }
  }

  // ---- layer 3: [64x3] too skinny for WMMA -> VALU per lane ---------------
  float o0 = sB3[0], o1 = sB3[1], o2 = sB3[2];
  const _Float16* hr = &Hbuf[lane * HID];
#pragma unroll
  for (int k = 0; k < HID; ++k) {
    const float hv = (float)hr[k];
    o0 += hv * sW3[k * 3 + 0];
    o1 += hv * sW3[k * 3 + 1];
    o2 += hv * sW3[k * 3 + 2];
  }
  out[p * 3 + 0] = x0 + o0;
  out[p * 3 + 1] = x1 + o1;
  out[p * 3 + 2] = x2 + o2;
}

// Scalar fallback for a non-multiple-of-256 tail (N=2^20 in the reference, so
// this normally never launches; kept so WMMA EXEC==all-ones is never violated).
__global__ void nesvor_tail(
    const float* __restrict__ x, const float* __restrict__ e,
    const float* __restrict__ tables,
    const float* __restrict__ W1, const float* __restrict__ b1,
    const float* __restrict__ W2, const float* __restrict__ b2,
    const float* __restrict__ W3, const float* __restrict__ b3,
    float* __restrict__ out, int start, int n) {
  const int p = start + blockIdx.x * 64 + threadIdx.x;
  if (p >= n) return;
  const float x0 = x[p * 3 + 0], x1 = x[p * 3 + 1], x2 = x[p * 3 + 2];
  float in[DIN];
  float rf = 16.0f;
  for (int lev = 0; lev < LVL; ++lev) {
    const float r = floorf(rf); rf *= 1.5f;
    const float px = x0 * r, py = x1 * r, pz = x2 * r;
    const float fx = floorf(px), fy = floorf(py), fz = floorf(pz);
    const unsigned ix = (unsigned)fx, iy = (unsigned)fy, iz = (unsigned)fz;
    const float tx = smoothstep_w(px - fx), ty = smoothstep_w(py - fy),
                tz = smoothstep_w(pz - fz);
    const float2* tab = reinterpret_cast<const float2*>(tables) + (size_t)lev * TSZ;
    float f0 = 0.0f, f1 = 0.0f;
    for (int c = 0; c < 8; ++c) {
      unsigned h = (ix + ((c >> 2) & 1)) ^ ((iy + ((c >> 1) & 1)) * 2654435761u) ^
                   ((iz + (c & 1)) * 805459861u);
      h &= (TSZ - 1u);
      const float2 f = tab[h];
      const float w = ((c & 4) ? tx : 1.0f - tx) * ((c & 2) ? ty : 1.0f - ty) *
                      ((c & 1) ? tz : 1.0f - tz);
      f0 += w * f.x; f1 += w * f.y;
    }
    in[2 * lev] = f0; in[2 * lev + 1] = f1;
  }
  for (int k = 0; k < 16; ++k) in[16 + k] = e[(size_t)p * 16 + k];
  float h1[HID], h2[HID];
  for (int j = 0; j < HID; ++j) {
    float s = b1[j];
    for (int k = 0; k < DIN; ++k) s += in[k] * W1[k * HID + j];
    h1[j] = fast_tanh(s);
  }
  for (int j = 0; j < HID; ++j) {
    float s = b2[j];
    for (int k = 0; k < HID; ++k) s += h1[k] * W2[k * HID + j];
    h2[j] = fast_tanh(s);
  }
  for (int j = 0; j < 3; ++j) {
    float s = b3[j];
    for (int k = 0; k < HID; ++k) s += h2[k] * W3[k * 3 + j];
    out[p * 3 + j] = x[p * 3 + j] + s;
  }
}

extern "C" void kernel_launch(void* const* d_in, const int* in_sizes, int n_in,
                              void* d_out, int out_size, void* d_ws, size_t ws_size,
                              hipStream_t stream) {
  const float* x      = (const float*)d_in[0];
  const float* e      = (const float*)d_in[1];
  const float* tables = (const float*)d_in[2];
  const float* W1     = (const float*)d_in[3];
  const float* b1     = (const float*)d_in[4];
  const float* W2     = (const float*)d_in[5];
  const float* b2     = (const float*)d_in[6];
  const float* W3     = (const float*)d_in[7];
  const float* b3     = (const float*)d_in[8];
  float* out = (float*)d_out;

  const int n      = in_sizes[0] / 3;   // number of points (2^20)
  const int blocks = n / 256;
  if (blocks > 0)
    nesvor_deform<<<blocks, 256, 0, stream>>>(x, e, tables, W1, b1, W2, b2, W3,
                                              b3, out);
  const int done = blocks * 256;
  const int rem  = n - done;
  if (rem > 0)
    nesvor_tail<<<(rem + 63) / 64, 64, 0, stream>>>(x, e, tables, W1, b1, W2,
                                                    b2, W3, b3, out, done, n);
}
